// DiscriminativeModel_72310069395532
// MI455X (gfx1250) — matrix operand order repaired and checked by reading, not compiled, not run on hardware
//
#include <hip/hip_runtime.h>
#include <hip/hip_bf16.h>
#include <math.h>

typedef __attribute__((ext_vector_type(16))) _Float16 v16h;
typedef __attribute__((ext_vector_type(8)))  float    v8f;

#define NUM_REL 8
#define DIM 128          // DIM_IN == HIDDEN == 128
#define KTOT (NUM_REL * DIM)   // 1024

// ---------------------------------------------------------------------------
// Pack a row-major f32 B matrix [krows x 128] into WMMA per-lane f16 order.
// Packed index: ((kt*8 + nt)*32 + lane)*16 + j
//   element = B[kt*32 + (lane&16 ? 8:0) + (j<8 ? j : j+8)][nt*16 + (lane&15)]
// ---------------------------------------------------------------------------
__global__ void pack_b_kernel(const float* __restrict__ src,
                              _Float16* __restrict__ dst, int krows) {
    int t = blockIdx.x * blockDim.x + threadIdx.x;
    int total = krows * 128;
    if (t >= total) return;
    int j    = t & 15;
    int lane = (t >> 4) & 31;
    int nt   = (t >> 9) & 7;
    int kt   = t >> 12;
    int K = kt * 32 + ((lane & 16) ? 8 : 0) + (j < 8 ? j : j + 8);
    int N = nt * 16 + (lane & 15);
    dst[t] = (_Float16)src[K * 128 + N];
}

// ---------------------------------------------------------------------------
// Per-(dst,rel) edge counts
// ---------------------------------------------------------------------------
__global__ void count_edges_kernel(const int* __restrict__ dst,
                                   const int* __restrict__ rel,
                                   int* __restrict__ cnt, int nedges) {
    int e = blockIdx.x * blockDim.x + threadIdx.x;
    if (e < nedges) atomicAdd(&cnt[dst[e] * NUM_REL + rel[e]], 1);
}

// ---------------------------------------------------------------------------
// Scatter normalized source features into s[(dst*8+rel)][0..127].
// One wave per edge; lane handles 4 contiguous floats (float4 read, 4 atomics).
// xidx != nullptr => source row = xbase[xidx[src]] (embedding gather, layer 1).
// ---------------------------------------------------------------------------
__global__ void scatter_edges_kernel(const int* __restrict__ srcIdx,
                                     const int* __restrict__ dstIdx,
                                     const int* __restrict__ rel,
                                     const int* __restrict__ cnt,
                                     const float* __restrict__ xbase,
                                     const int* __restrict__ xidx,
                                     float* __restrict__ s, int nedges) {
    int e    = blockIdx.x * (blockDim.x >> 5) + (threadIdx.x >> 5);
    int lane = threadIdx.x & 31;
    if (e >= nedges) return;
    int sN = srcIdx[e];
    int comp = dstIdx[e] * NUM_REL + rel[e];
    int c = cnt[comp];
    float norm = 1.0f / (float)(c > 1 ? c : 1);
    int row = xidx ? xidx[sN] : sN;
    const float* xrow = xbase + (size_t)row * DIM;
    float* srow = s + (size_t)comp * DIM;
    float4 v = ((const float4*)xrow)[lane];
    atomicAdd(&srow[lane * 4 + 0], v.x * norm);
    atomicAdd(&srow[lane * 4 + 1], v.y * norm);
    atomicAdd(&srow[lane * 4 + 2], v.z * norm);
    atomicAdd(&srow[lane * 4 + 3], v.w * norm);
}

// ---------------------------------------------------------------------------
// A-fragment: 16x32 f16 tile per the CDNA5 ISA layout, built from a row-major
// f32 row (K contiguous).  lane<16: K = kbase+0..7, kbase+16..23 of row M=lane
//                          lane>=16: K = kbase+8..15, kbase+24..31 (same row).
// ---------------------------------------------------------------------------
__device__ __forceinline__ v16h load_a_frag(const float* __restrict__ rowPtr,
                                            int kbase, int lane) {
    const float4* p = (const float4*)(rowPtr + kbase + ((lane & 16) ? 8 : 0));
    float4 x0 = p[0], x1 = p[1];   // K + 0..7
    float4 y0 = p[4], y1 = p[5];   // K + 16..23
    v16h a;
    a[0]  = (_Float16)x0.x; a[1]  = (_Float16)x0.y;
    a[2]  = (_Float16)x0.z; a[3]  = (_Float16)x0.w;
    a[4]  = (_Float16)x1.x; a[5]  = (_Float16)x1.y;
    a[6]  = (_Float16)x1.z; a[7]  = (_Float16)x1.w;
    a[8]  = (_Float16)y0.x; a[9]  = (_Float16)y0.y;
    a[10] = (_Float16)y0.z; a[11] = (_Float16)y0.w;
    a[12] = (_Float16)y1.x; a[13] = (_Float16)y1.y;
    a[14] = (_Float16)y1.z; a[15] = (_Float16)y1.w;
    return a;
}

// ---------------------------------------------------------------------------
// Fused RGCN layer GEMM:
//   h[n,:] = relu( s[n, 0..1023] @ pB  +  x(n) @ pR  + bias )
// x(n) = xbase[xidx[n]] (layer 1: emb gather) or xbase[n] (layer 2).
// Block = 128 threads = 4 waves; each wave owns one 16-row M tile and all
// 8 N tiles.  C accumulators live in 8 x v8f.
// ---------------------------------------------------------------------------
__global__ void rgcn_gemm_kernel(const float* __restrict__ s,
                                 const _Float16* __restrict__ pB,   // packed [32*8*32*16]
                                 const float* __restrict__ xbase,
                                 const int* __restrict__ xidx,
                                 const _Float16* __restrict__ pR,   // packed [4*8*32*16]
                                 const float* __restrict__ bias,
                                 float* __restrict__ out,
                                 int nrows) {
    int wave  = threadIdx.x >> 5;
    int lane  = threadIdx.x & 31;
    int mtile = blockIdx.x * 4 + wave;
    int mtiles = nrows >> 4;              // nrows % 16 == 0 (50000 = 3125*16)
    if (mtile >= mtiles) return;          // wave-uniform: EXEC all-1s past here

    int m = mtile * 16 + (lane & 15);     // A row handled by this lane

    v8f c[8];
    #pragma unroll
    for (int nt = 0; nt < 8; ++nt)
        #pragma unroll
        for (int g = 0; g < 8; ++g) c[nt][g] = 0.0f;

    // --- neighbor-aggregate part: K = 1024 over s ---
    const float* arow = s + (size_t)m * KTOT;
    for (int kt = 0; kt < KTOT / 32; ++kt) {
        v16h a = load_a_frag(arow, kt * 32, lane);
        #pragma unroll
        for (int nt = 0; nt < 8; ++nt) {
            v16h b = *(const v16h*)(pB + (((size_t)(kt * 8 + nt) * 32 + lane) * 16));
            c[nt] = __builtin_amdgcn_wmma_f32_16x16x32_f16(
                false, a, false, b, (short)0, c[nt], false, false);
        }
    }

    // --- root part: K = 128 over x ---
    int xrowi = xidx ? xidx[m] : m;
    const float* xrow = xbase + (size_t)xrowi * DIM;
    for (int kt = 0; kt < DIM / 32; ++kt) {
        v16h a = load_a_frag(xrow, kt * 32, lane);
        #pragma unroll
        for (int nt = 0; nt < 8; ++nt) {
            v16h b = *(const v16h*)(pR + (((size_t)(kt * 8 + nt) * 32 + lane) * 16));
            c[nt] = __builtin_amdgcn_wmma_f32_16x16x32_f16(
                false, a, false, b, (short)0, c[nt], false, false);
        }
    }

    // --- bias + relu + store.  D layout: lane<16: N=lane, M=g; lane>=16: M=g+8.
    #pragma unroll
    for (int nt = 0; nt < 8; ++nt) {
        int ncol = nt * 16 + (lane & 15);
        float bv = bias[ncol];
        #pragma unroll
        for (int g = 0; g < 8; ++g) {
            int Mrow = mtile * 16 + ((lane & 16) ? (g + 8) : g);
            float v = c[nt][g] + bv;
            out[(size_t)Mrow * DIM + ncol] = v > 0.0f ? v : 0.0f;
        }
    }
}

// ---------------------------------------------------------------------------
// scores[n] = h[n,:] . att_v   (one wave per node, shuffle reduction)
// ---------------------------------------------------------------------------
__global__ void scores_kernel(const float* __restrict__ h,
                              const float* __restrict__ att,
                              float* __restrict__ scores, int n) {
    int node = blockIdx.x * (blockDim.x >> 5) + (threadIdx.x >> 5);
    int lane = threadIdx.x & 31;
    if (node >= n) return;
    const float* row = h + (size_t)node * DIM;
    float acc = 0.0f;
    #pragma unroll
    for (int k = 0; k < DIM / 32; ++k) acc += row[lane + 32 * k] * att[lane + 32 * k];
    #pragma unroll
    for (int o = 16; o > 0; o >>= 1) acc += __shfl_down(acc, o, 32);
    if (lane == 0) scores[node] = acc;
}

// ---------------------------------------------------------------------------
// Segment boundaries from sorted batch vector: gstart[g]..gstart[g+1])
// ---------------------------------------------------------------------------
__global__ void seg_starts_kernel(const int* __restrict__ bs,
                                  int* __restrict__ gstart, int n, int ngraph) {
    int i = blockIdx.x * blockDim.x + threadIdx.x;
    if (i >= n) return;
    int cur = bs[i];
    int prev = (i == 0) ? -1 : bs[i - 1];
    if (cur != prev)
        for (int g = prev + 1; g <= cur; ++g) gstart[g] = i;
    if (i == n - 1)
        for (int g = cur + 1; g <= ngraph; ++g) gstart[g] = n;
}

// ---------------------------------------------------------------------------
// Per-graph softmax attention pooling + final linear + sigmoid.
// 64 blocks x 128 threads; thread t owns feature t.
// ---------------------------------------------------------------------------
__global__ void pool_kernel(const float* __restrict__ h,
                            const float* __restrict__ scores,
                            const int* __restrict__ gstart,
                            const float* __restrict__ lin_w,
                            const float* __restrict__ lin_b,
                            float* __restrict__ out) {
    __shared__ float red[128];
    __shared__ float wbuf[128];
    int g = blockIdx.x, t = threadIdx.x;
    int beg = gstart[g], end = gstart[g + 1];

    // segment max
    float m = -3.4e38f;
    for (int i = beg + t; i < end; i += 128) m = fmaxf(m, scores[i]);
    red[t] = m; __syncthreads();
    for (int o = 64; o > 0; o >>= 1) {
        if (t < o) red[t] = fmaxf(red[t], red[t + o]);
        __syncthreads();
    }
    m = red[0]; __syncthreads();

    // segment sum of exp
    float se = 0.0f;
    for (int i = beg + t; i < end; i += 128) se += __expf(scores[i] - m);
    red[t] = se; __syncthreads();
    for (int o = 64; o > 0; o >>= 1) {
        if (t < o) red[t] += red[t + o];
        __syncthreads();
    }
    float denom = red[0]; __syncthreads();
    float inv = (denom > 0.0f) ? 1.0f / denom : 0.0f;

    // weighted feature sum: graph_emb[t]
    float acc = 0.0f;
    for (int base = beg; base < end; base += 128) {
        int i = base + t;
        wbuf[t] = (i < end) ? __expf(scores[i] - m) * inv : 0.0f;
        __syncthreads();
        int lim = end - base; if (lim > 128) lim = 128;
        for (int j = 0; j < lim; ++j)
            acc += wbuf[j] * h[(size_t)(base + j) * DIM + t];
        __syncthreads();
    }

    // dot(graph_emb, lin_w) + lin_b, sigmoid
    red[t] = acc * lin_w[t]; __syncthreads();
    for (int o = 64; o > 0; o >>= 1) {
        if (t < o) red[t] += red[t + o];
        __syncthreads();
    }
    if (t == 0) {
        float z = red[0] + lin_b[0];
        out[g] = 1.0f / (1.0f + __expf(-z));
    }
}

// ---------------------------------------------------------------------------
extern "C" void kernel_launch(void* const* d_in, const int* in_sizes, int n_in,
                              void* d_out, int out_size, void* d_ws, size_t ws_size,
                              hipStream_t stream) {
    const int*   nodeTypes = (const int*)  d_in[0];
    const int*   edge_idx  = (const int*)  d_in[1];   // [2, E]
    const int*   edge_attr = (const int*)  d_in[2];
    const int*   bs        = (const int*)  d_in[3];
    const float* emb       = (const float*)d_in[4];
    const float* W1        = (const float*)d_in[5];
    const float* root1     = (const float*)d_in[6];
    const float* b1        = (const float*)d_in[7];
    const float* W2        = (const float*)d_in[8];
    const float* root2     = (const float*)d_in[9];
    const float* b2        = (const float*)d_in[10];
    const float* att_v     = (const float*)d_in[11];
    const float* lin_w     = (const float*)d_in[12];
    const float* lin_b     = (const float*)d_in[13];
    float*       out       = (float*)d_out;

    const int N = in_sizes[0];            // 50000
    const int E = in_sizes[2];            // 800000
    const int G = out_size;               // 64
    const int* srcIdx = edge_idx;
    const int* dstIdx = edge_idx + E;

    // --- workspace carve (chunks 256B-aligned for N = 50000, except gstart) ---
    char* w = (char*)d_ws;
    float*     sbuf   = (float*)w;     w += (size_t)N * NUM_REL * DIM * sizeof(float); // 204.8 MB
    int*       cnt    = (int*)w;       w += (size_t)N * NUM_REL * sizeof(int);         // 1.6 MB
    float*     h1     = (float*)w;     w += (size_t)N * DIM * sizeof(float);           // 25.6 MB
    float*     h2     = (float*)w;     w += (size_t)N * DIM * sizeof(float);           // 25.6 MB
    _Float16*  pB1    = (_Float16*)w;  w += (size_t)KTOT * DIM * sizeof(_Float16);     // 256 KB
    _Float16*  pR1    = (_Float16*)w;  w += (size_t)DIM * DIM * sizeof(_Float16);      // 32 KB
    _Float16*  pB2    = (_Float16*)w;  w += (size_t)KTOT * DIM * sizeof(_Float16);
    _Float16*  pR2    = (_Float16*)w;  w += (size_t)DIM * DIM * sizeof(_Float16);
    float*     scores = (float*)w;     w += (size_t)N * sizeof(float);                 // 200 KB
    int*       gstart = (int*)w;       w += (size_t)(G + 1) * sizeof(int);

    const size_t sBytes = (size_t)N * NUM_REL * DIM * sizeof(float);

    // --- init ---
    hipMemsetAsync(cnt, 0, (size_t)N * NUM_REL * sizeof(int), stream);
    hipMemsetAsync(sbuf, 0, sBytes, stream);

    // --- pack weights to WMMA per-lane f16 order (tiny) ---
    pack_b_kernel<<<(KTOT * DIM + 255) / 256, 256, 0, stream>>>(W1, pB1, KTOT);
    pack_b_kernel<<<(DIM * DIM + 255) / 256, 256, 0, stream>>>(root1, pR1, DIM);
    pack_b_kernel<<<(KTOT * DIM + 255) / 256, 256, 0, stream>>>(W2, pB2, KTOT);
    pack_b_kernel<<<(DIM * DIM + 255) / 256, 256, 0, stream>>>(root2, pR2, DIM);

    // --- per-(dst,rel) counts ---
    count_edges_kernel<<<(E + 255) / 256, 256, 0, stream>>>(dstIdx, edge_attr, cnt, E);

    // --- layer 1: scatter emb[nodeTypes[src]] -> s, then fused GEMM -> h1 ---
    scatter_edges_kernel<<<(E + 7) / 8, 256, 0, stream>>>(
        srcIdx, dstIdx, edge_attr, cnt, emb, nodeTypes, sbuf, E);
    {
        int mtiles = N >> 4;                       // 3125
        rgcn_gemm_kernel<<<(mtiles + 3) / 4, 128, 0, stream>>>(
            sbuf, pB1, emb, nodeTypes, pR1, b1, h1, N);
    }

    // --- layer 2: re-zero s, scatter h1[src] -> s, fused GEMM -> h2 ---
    hipMemsetAsync(sbuf, 0, sBytes, stream);
    scatter_edges_kernel<<<(E + 7) / 8, 256, 0, stream>>>(
        srcIdx, dstIdx, edge_attr, cnt, h1, nullptr, sbuf, E);
    {
        int mtiles = N >> 4;
        rgcn_gemm_kernel<<<(mtiles + 3) / 4, 128, 0, stream>>>(
            sbuf, pB2, h1, nullptr, pR2, b2, h2, N);
    }

    // --- attention pooling ---
    scores_kernel<<<(N + 7) / 8, 256, 0, stream>>>(h2, att_v, scores, N);
    seg_starts_kernel<<<(N + 255) / 256, 256, 0, stream>>>(bs, gstart, N, G);
    pool_kernel<<<G, 128, 0, stream>>>(h2, scores, gstart, lin_w, lin_b, out);
}
